// SelfAttention_9045201126021
// MI455X (gfx1250) — matrix operand order repaired and checked by reading, not compiled, not run on hardware
//
#include <hip/hip_runtime.h>
#include <hip/hip_bf16.h>

// ---------------- problem constants ----------------
#define B_   8
#define C_   512
#define T_   1024
#define HEADS_ 8
#define CH_  64
#define GROUPS_ 32
#define CPG_ 16
#define M3C_ 1536
#define EPS_ 1e-5f
#define RES_SCALE_ 0.7071067811865476f
#define SCALE_QK_  0.3535533905932738f   // 64^-0.25

typedef __attribute__((ext_vector_type(16))) _Float16 v16h;
typedef __attribute__((ext_vector_type(8)))  float    v8f;
typedef __attribute__((ext_vector_type(4)))  unsigned int v4u;
typedef __attribute__((ext_vector_type(8)))  int      v8i;
typedef __attribute__((ext_vector_type(4)))  int      v4i;

union U8H { uint4 v; _Float16 h[8]; };

// Tensor Data Mover availability (this toolchain: 6-arg builtin form)
#if defined(__has_builtin)
#  if __has_builtin(__builtin_amdgcn_tensor_load_to_lds) && \
      __has_builtin(__builtin_amdgcn_s_wait_tensorcnt)
#    define USE_TDM 1
#  endif
#endif
#ifndef USE_TDM
#  define USE_TDM 0
#endif

// ---------------- WMMA helper (CDNA5 16x16x32 f16 -> f32) ----------------
__device__ __forceinline__ v8f wmma16(v16h a, v16h b, v8f c) {
    return __builtin_amdgcn_wmma_f32_16x16x32_f16(false, a, false, b, (short)0, c, false, false);
}

// A fragment: row-major [16][K>=32] tile, ld in halfs.
__device__ __forceinline__ v16h lda_frag(const _Float16* base, int ld) {
    const int lane = threadIdx.x & 31;
    const int m = lane & 15, h = lane >> 4;
    const _Float16* p0 = base + m * ld + h * 16;
    const _Float16* p1 = base + m * ld + h * 16 + 8;
    v16h a;
#pragma unroll
    for (int e = 0; e < 8; ++e) { a[e] = p0[e]; a[8 + e] = p1[e]; }
    return a;
}

// B fragment from transposed tile Bt[n][k] (ld in halfs).
__device__ __forceinline__ v16h ldbT_frag(const _Float16* base, int ld, int koff) {
    const int lane = threadIdx.x & 31;
    const int n = lane & 15, h = lane >> 4;
    const _Float16* p = base + n * ld + koff + h * 16;
    v16h b;
#pragma unroll
    for (int e = 0; e < 16; ++e) b[e] = p[e];
    return b;
}

#if USE_TDM
// TDM 2D tile load: 64 rows x 64 halfs, row stride 1024 halfs, into LDS rows
// padded to 72 halfs (pad 4 DWORDs after every 32 DWORDs = one 128B row).
__device__ __forceinline__ void tdm_load_2d(const _Float16* gsrc, _Float16* lds_dst) {
    unsigned long long ga = (unsigned long long)(size_t)gsrc;
    v4u g0;
    g0[0] = 1u;                                              // count=1, user mode
    g0[1] = (unsigned)(size_t)lds_dst;                       // LDS byte address
    g0[2] = (unsigned)ga;                                    // global_addr[31:0]
    g0[3] = (unsigned)((ga >> 32) & 0x1FFFFFFu) | (2u << 30); // addr[56:32] | type=2
    v8i g1;
    g1[0] = 0x07110000;        // data_size=2B | pad_enable | pad_interval=32dw | pad_amount=4dw
    g1[1] = (int)(64u << 16);  // tensor_dim0 = 64
    g1[2] = (int)(64u << 16);  // tensor_dim1 = 64
    g1[3] = (int)(64u << 16);  // tile_dim0   = 64
    g1[4] = 64;                // tile_dim1   = 64
    g1[5] = 1024;              // tensor_dim0_stride = 1024 elements
    g1[6] = 0;
    g1[7] = 0;
    v4i z4 = {0, 0, 0, 0};
    v8i z8 = {0, 0, 0, 0, 0, 0, 0, 0};
    __builtin_amdgcn_tensor_load_to_lds(g0, g1, z4, z4, z8, 0);
}
#endif

// ---------------- weight conversion f32 -> f16 ----------------
__global__ void __launch_bounds__(256) k_cvt(const float* __restrict__ wq,
                                             const float* __restrict__ wp,
                                             _Float16* __restrict__ wqh,
                                             _Float16* __restrict__ wph) {
    const int stride = gridDim.x * blockDim.x;
    const int gid = blockIdx.x * blockDim.x + threadIdx.x;
    for (int i = gid; i < M3C_ * C_; i += stride) wqh[i] = (_Float16)wq[i];
    for (int i = gid; i < C_ * C_;  i += stride) wph[i] = (_Float16)wp[i];
}

// ---------------- GroupNorm (affine) -> xn f32 ----------------
__global__ void __launch_bounds__(256) k_gn1(const float* __restrict__ x,
                                             const float* __restrict__ gw,
                                             const float* __restrict__ gb,
                                             float* __restrict__ xn) {
    const int bi = blockIdx.x >> 5;
    const int g  = blockIdx.x & 31;
    const int tid = threadIdx.x;
    const int N = CPG_ * T_;
    const size_t off = ((size_t)bi * C_ + g * CPG_) * T_;
    const float* xp = x + off;

    float s = 0.f, sq = 0.f;
    for (int i = tid; i < N; i += 256) { float v = xp[i]; s += v; sq += v * v; }
    __shared__ float s1[256], s2[256];
    s1[tid] = s; s2[tid] = sq; __syncthreads();
    for (int w = 128; w > 0; w >>= 1) {
        if (tid < w) { s1[tid] += s1[tid + w]; s2[tid] += s2[tid + w]; }
        __syncthreads();
    }
    const float mean = s1[0] * (1.0f / N);
    const float var  = s2[0] * (1.0f / N) - mean * mean;
    const float inv  = rsqrtf(var + EPS_);
    for (int i = tid; i < N; i += 256) {
        int ch = g * CPG_ + (i >> 10);
        xn[off + i] = (xp[i] - mean) * inv * gw[ch] + gb[ch];
    }
}

// ---------------- transpose xn[c][t] (f32) -> xnhT[t][c] (f16) -------------
__global__ void __launch_bounds__(256) k_tr(const float* __restrict__ xn,
                                            _Float16* __restrict__ xt) {
    const int bi = blockIdx.z;
    const int c0 = blockIdx.y * 64;
    const int t0 = blockIdx.x * 64;
    const int tid = threadIdx.x;
    __shared__ __align__(16) _Float16 tile[64][72];

    {
        int rr = tid >> 2, cc = (tid & 3) * 16;
        const float* s = xn + ((size_t)bi * C_ + c0 + rr) * T_ + t0 + cc;
#pragma unroll
        for (int e = 0; e < 16; ++e) tile[rr][cc + e] = (_Float16)s[e];
    }
    __syncthreads();
    {
        int tt = tid >> 2, c2 = (tid & 3) * 16;
        U8H u0, u1;
#pragma unroll
        for (int e = 0; e < 8; ++e) { u0.h[e] = tile[c2 + e][tt]; u1.h[e] = tile[c2 + 8 + e][tt]; }
        _Float16* d = xt + ((size_t)bi * T_ + t0 + tt) * C_ + c0 + c2;
        ((uint4*)d)[0] = u0.v;
        ((uint4*)d)[1] = u1.v;
    }
}

// ---------------- 128x64 GEMM core: W[m][k] x XT[n][k]^T, reg dbl-buffer ---
__device__ __forceinline__ void gemm_128x64(const _Float16* __restrict__ Wrow,
                                            const _Float16* __restrict__ XTrow,
                                            int ldxt, v8f (&acc)[2][2]) {
    __shared__ __align__(16) _Float16 As[128][32];
    __shared__ __align__(16) _Float16 Bs[64][40];
    const int tid  = threadIdx.x;
    const int wave = tid >> 5;
    const int mt0  = (wave & 3) * 2;
    const int nt   = (wave >> 2) * 2;

    const int ra = tid >> 1, ca = (tid & 1) * 16;
    const int rb = tid >> 2, cb = (tid & 3) * 8;

    uint4 rega0, rega1, regb;
    {
        const _Float16* aw = Wrow + (size_t)ra * C_ + ca;
        rega0 = *(const uint4*)(aw);
        rega1 = *(const uint4*)(aw + 8);
        regb  = *(const uint4*)(XTrow + (size_t)rb * ldxt + cb);
    }
    for (int k0 = 0; k0 < C_; k0 += 32) {
        *(uint4*)&As[ra][ca]     = rega0;
        *(uint4*)&As[ra][ca + 8] = rega1;
        *(uint4*)&Bs[rb][cb]     = regb;
        __syncthreads();
        if (k0 + 32 < C_) {
            const _Float16* aw = Wrow + (size_t)ra * C_ + k0 + 32 + ca;
            rega0 = *(const uint4*)(aw);
            rega1 = *(const uint4*)(aw + 8);
            regb  = *(const uint4*)(XTrow + (size_t)rb * ldxt + k0 + 32 + cb);
        }
        v16h a0 = lda_frag(&As[(mt0 + 0) * 16][0], 32);
        v16h a1 = lda_frag(&As[(mt0 + 1) * 16][0], 32);
        v16h b0 = ldbT_frag(&Bs[(nt + 0) * 16][0], 40, 0);
        v16h b1 = ldbT_frag(&Bs[(nt + 1) * 16][0], 40, 0);
        acc[0][0] = wmma16(a0, b0, acc[0][0]);
        acc[0][1] = wmma16(a0, b1, acc[0][1]);
        acc[1][0] = wmma16(a1, b0, acc[1][0]);
        acc[1][1] = wmma16(a1, b1, acc[1][1]);
        __syncthreads();
    }
}

// ---------------- QKV GEMM: Q/K -> t-major qkT, V -> c-major vh ------------
__global__ void __launch_bounds__(256) k_qkv(const _Float16* __restrict__ Wh,
                                             const _Float16* __restrict__ xt,
                                             const float* __restrict__ bias,
                                             _Float16* __restrict__ qkT,
                                             _Float16* __restrict__ vh) {
    const int bi = blockIdx.z;
    const int n0 = blockIdx.x * 64;
    const int m0 = blockIdx.y * 128;
    v8f acc[2][2];
#pragma unroll
    for (int i = 0; i < 2; ++i)
#pragma unroll
        for (int j = 0; j < 2; ++j) acc[i][j] = (v8f){};

    gemm_128x64(Wh + (size_t)m0 * C_, xt + ((size_t)bi * T_ + n0) * C_, C_, acc);

    const int tid = threadIdx.x, wave = tid >> 5, lane = tid & 31;
    const int mt0 = (wave & 3) * 2, nt = (wave >> 2) * 2;
    const int nn = lane & 15, mb = (lane >> 4) * 8;
#pragma unroll
    for (int i = 0; i < 2; ++i)
#pragma unroll
        for (int j = 0; j < 2; ++j) {
            const int col  = n0 + (nt + j) * 16 + nn;
            const int row0 = m0 + (mt0 + i) * 16 + mb;
            const int head0 = row0 / 192;
            const int w0 = row0 - head0 * 192;
            if (w0 < 128) {
                U8H u;
#pragma unroll
                for (int r = 0; r < 8; ++r)
                    u.h[r] = (_Float16)((acc[i][j][r] + bias[row0 + r]) * SCALE_QK_);
                *(uint4*)(qkT + ((size_t)bi * T_ + col) * 1024 + head0 * 128 + w0) = u.v;
            } else {
#pragma unroll
                for (int r = 0; r < 8; ++r)
                    vh[((size_t)bi * C_ + head0 * 64 + (w0 - 128) + r) * T_ + col] =
                        (_Float16)(acc[i][j][r] + bias[row0 + r]);
            }
        }
}

// ---------------- Flash attention (S^T = K^T x Q), t-major output ----------
__global__ void __launch_bounds__(128) k_attn(const _Float16* __restrict__ qkT,
                                              const _Float16* __restrict__ vh,
                                              _Float16* __restrict__ aT) {
    const int hb = blockIdx.y;
    const int t0 = blockIdx.x * 64;
    const int bi = hb >> 3;
    const int head = hb & 7;

    const int tid = threadIdx.x;
    const int wave = tid >> 5;
    const int lane = tid & 31;
    const int nn = lane & 15, h = lane >> 4;

    __shared__ __align__(16) _Float16 qs[64][72];     // Q^T [t][ch]
    __shared__ __align__(16) _Float16 ps[4][16][40];  // per-wave P [t][s(32)]

    const int cprow = tid >> 1, cpc = (tid & 1) * 32;
    {   // Q^T rows straight from qkT (vector copy)
        const uint4* s = (const uint4*)(qkT + ((size_t)bi * T_ + t0 + cprow) * 1024
                                        + head * 128 + cpc);
        uint4* d = (uint4*)&qs[cprow][cpc];
#pragma unroll
        for (int e = 0; e < 4; ++e) d[e] = s[e];
    }
    __syncthreads();
    const v16h bq0 = ldbT_frag(&qs[wave * 16][0], 72, 0);
    const v16h bq1 = ldbT_frag(&qs[wave * 16][0], 72, 32);

    float run_m = -3.0e38f, run_s = 0.f;
    v8f out_acc[4];
#pragma unroll
    for (int i = 0; i < 4; ++i) out_acc[i] = (v8f){};

    const _Float16* Ksrc = qkT + (size_t)bi * T_ * 1024 + head * 128 + 64;  // + s*1024
    const _Float16* Vsrc = vh + ((size_t)bi * C_ + head * 64) * T_;         // + ch*1024 + s

#if USE_TDM
    // --- Tensor Data Mover double-buffered staging ---
    __shared__ __align__(16) _Float16 kt[2][64][72];
    __shared__ __align__(16) _Float16 vt[2][64][72];
    if (wave == 0) {
        tdm_load_2d(Ksrc, &kt[0][0][0]);
        tdm_load_2d(Vsrc, &vt[0][0][0]);
    }
    int cur = 0;
    for (int s0 = 0; s0 < T_; s0 += 64, cur ^= 1) {
        if (wave == 0) __builtin_amdgcn_s_wait_tensorcnt(0);
        __syncthreads();   // publish DMA data; all waves done with prev buffer
        if (wave == 0 && s0 + 64 < T_) {
            tdm_load_2d(Ksrc + (size_t)(s0 + 64) * 1024, &kt[cur ^ 1][0][0]);
            tdm_load_2d(Vsrc + (s0 + 64), &vt[cur ^ 1][0][0]);
        }
        const _Float16* ktc = &kt[cur][0][0];
        const _Float16* vtc = &vt[cur][0][0];
#else
    // --- manual register double-buffered staging ---
    __shared__ __align__(16) _Float16 kt[64][72];
    __shared__ __align__(16) _Float16 vt[64][72];
    uint4 kreg[4], vreg[4];
    {
        const uint4* ks = (const uint4*)(Ksrc + (size_t)cprow * 1024 + cpc);
        const uint4* vs = (const uint4*)(Vsrc + (size_t)cprow * T_ + cpc);
#pragma unroll
        for (int e = 0; e < 4; ++e) { kreg[e] = ks[e]; vreg[e] = vs[e]; }
    }
    for (int s0 = 0; s0 < T_; s0 += 64) {
        __syncthreads();
        {
            uint4* kd = (uint4*)&kt[cprow][cpc];
            uint4* vd = (uint4*)&vt[cprow][cpc];
#pragma unroll
            for (int e = 0; e < 4; ++e) { kd[e] = kreg[e]; vd[e] = vreg[e]; }
        }
        __syncthreads();
        if (s0 + 64 < T_) {
            const uint4* ks = (const uint4*)(Ksrc + (size_t)(s0 + 64 + cprow) * 1024 + cpc);
            const uint4* vs = (const uint4*)(Vsrc + (size_t)cprow * T_ + s0 + 64 + cpc);
#pragma unroll
            for (int e = 0; e < 4; ++e) { kreg[e] = ks[e]; vreg[e] = vs[e]; }
        }
        const _Float16* ktc = &kt[0][0];
        const _Float16* vtc = &vt[0][0];
#endif

#pragma unroll
        for (int sh = 0; sh < 2; ++sh) {
            const int sb = sh * 32;
            v16h ka0  = lda_frag(ktc + sb * 72, 72);
            v16h ka0b = lda_frag(ktc + sb * 72 + 32, 72);
            v16h ka1  = lda_frag(ktc + (sb + 16) * 72, 72);
            v16h ka1b = lda_frag(ktc + (sb + 16) * 72 + 32, 72);
            v8f sA = {}; sA = wmma16(ka0,  bq0, sA); sA = wmma16(ka0b, bq1, sA);
            v8f sB = {}; sB = wmma16(ka1,  bq0, sB); sB = wmma16(ka1b, bq1, sB);

            float tm = fmaxf(sA[0], sB[0]);
#pragma unroll
            for (int r = 1; r < 8; ++r) tm = fmaxf(tm, fmaxf(sA[r], sB[r]));
            tm = fmaxf(tm, __shfl_xor(tm, 16, 32));
            const float nm = fmaxf(run_m, tm);
            const float corr = __expf(run_m - nm);
            float ts = 0.f;
#pragma unroll
            for (int r = 0; r < 8; ++r) {
                float ea = __expf(sA[r] - nm);
                float eb = __expf(sB[r] - nm);
                ps[wave][nn][h * 8 + r]      = (_Float16)ea;
                ps[wave][nn][16 + h * 8 + r] = (_Float16)eb;
                ts += ea + eb;
            }
            ts += __shfl_xor(ts, 16, 32);
            run_s = run_s * corr + ts;
            run_m = nm;
#pragma unroll
            for (int mt = 0; mt < 4; ++mt)
#pragma unroll
                for (int r = 0; r < 8; ++r) out_acc[mt][r] *= corr;

            v16h bp = ldbT_frag(&ps[wave][0][0], 40, 0);
#pragma unroll
            for (int mt = 0; mt < 4; ++mt) {
                v16h av = lda_frag(vtc + (mt * 16) * 72 + sb, 72);
                out_acc[mt] = wmma16(av, bp, out_acc[mt]);
            }
        }
    }

    const float inv = 1.0f / run_s;
    const int t = t0 + wave * 16 + nn;
    _Float16* dst = aT + ((size_t)bi * T_ + t) * C_ + head * 64 + h * 8;
#pragma unroll
    for (int mt = 0; mt < 4; ++mt) {
        U8H u;
#pragma unroll
        for (int r = 0; r < 8; ++r) u.h[r] = (_Float16)(out_acc[mt][r] * inv);
        *(uint4*)(dst + mt * 16) = u.v;
    }
}

// ---------------- proj GEMM + residual ----------------
__global__ void __launch_bounds__(256) k_proj(const _Float16* __restrict__ Wh,
                                              const _Float16* __restrict__ aT,
                                              const float* __restrict__ bias,
                                              const float* __restrict__ xn,
                                              float* __restrict__ Rf) {
    const int bi = blockIdx.z;
    const int n0 = blockIdx.x * 64;
    const int m0 = blockIdx.y * 128;
    v8f acc[2][2];
#pragma unroll
    for (int i = 0; i < 2; ++i)
#pragma unroll
        for (int j = 0; j < 2; ++j) acc[i][j] = (v8f){};

    gemm_128x64(Wh + (size_t)m0 * C_, aT + ((size_t)bi * T_ + n0) * C_, C_, acc);

    const int tid = threadIdx.x, wave = tid >> 5, lane = tid & 31;
    const int mt0 = (wave & 3) * 2, nt = (wave >> 2) * 2;
    const int nn = lane & 15, mb = (lane >> 4) * 8;
#pragma unroll
    for (int i = 0; i < 2; ++i)
#pragma unroll
        for (int j = 0; j < 2; ++j) {
            int col = n0 + (nt + j) * 16 + nn;
#pragma unroll
            for (int r = 0; r < 8; ++r) {
                int row = m0 + (mt0 + i) * 16 + mb + r;
                size_t idx = ((size_t)bi * C_ + row) * T_ + col;
                Rf[idx] = RES_SCALE_ * xn[idx] + (acc[i][j][r] + bias[row]);
            }
        }
}

// ---------------- final GroupNorm (no affine) -> out ----------------------
__global__ void __launch_bounds__(256) k_gn2(const float* __restrict__ r,
                                             float* __restrict__ out) {
    const int bi = blockIdx.x >> 5;
    const int g  = blockIdx.x & 31;
    const int tid = threadIdx.x;
    const int N = CPG_ * T_;
    const size_t off = ((size_t)bi * C_ + g * CPG_) * T_;
    const float* xp = r + off;

    float s = 0.f, sq = 0.f;
    for (int i = tid; i < N; i += 256) { float v = xp[i]; s += v; sq += v * v; }
    __shared__ float s1[256], s2[256];
    s1[tid] = s; s2[tid] = sq; __syncthreads();
    for (int w = 128; w > 0; w >>= 1) {
        if (tid < w) { s1[tid] += s1[tid + w]; s2[tid] += s2[tid + w]; }
        __syncthreads();
    }
    const float mean = s1[0] * (1.0f / N);
    const float var  = s2[0] * (1.0f / N) - mean * mean;
    const float inv  = rsqrtf(var + EPS_);
    for (int i = tid; i < N; i += 256) out[off + i] = (xp[i] - mean) * inv;
}

// ---------------- host launcher -------------------------------------------
extern "C" void kernel_launch(void* const* d_in, const int* in_sizes, int n_in,
                              void* d_out, int out_size, void* d_ws, size_t ws_size,
                              hipStream_t stream) {
    (void)in_sizes; (void)n_in; (void)out_size; (void)ws_size;
    const float* x      = (const float*)d_in[0];
    const float* gn_w   = (const float*)d_in[1];
    const float* gn_b   = (const float*)d_in[2];
    const float* qkv_w  = (const float*)d_in[3];
    const float* qkv_b  = (const float*)d_in[4];
    const float* proj_w = (const float*)d_in[5];
    const float* proj_b = (const float*)d_in[6];
    float* out = (float*)d_out;

    const size_t NTOT = (size_t)B_ * C_ * T_;
    char* ws = (char*)d_ws;
    size_t off = 0;
    auto alloc = [&](size_t bytes) {
        void* p = ws + off;
        off += (bytes + 255) & ~(size_t)255;
        return p;
    };
    float*    xn   = (float*)alloc(NTOT * 4);
    float*    rf   = (float*)alloc(NTOT * 4);
    _Float16* xt   = (_Float16*)alloc(NTOT * 2);
    _Float16* qkT  = (_Float16*)alloc((size_t)B_ * T_ * 1024 * 2);
    _Float16* vh   = (_Float16*)alloc(NTOT * 2);
    _Float16* aT   = (_Float16*)alloc(NTOT * 2);
    _Float16* wqh  = (_Float16*)alloc((size_t)M3C_ * C_ * 2);
    _Float16* wph  = (_Float16*)alloc((size_t)C_ * C_ * 2);

    k_cvt<<<512, 256, 0, stream>>>(qkv_w, proj_w, wqh, wph);
    k_gn1<<<B_ * GROUPS_, 256, 0, stream>>>(x, gn_w, gn_b, xn);

    dim3 gt(T_ / 64, C_ / 64, B_);
    k_tr<<<gt, 256, 0, stream>>>(xn, xt);

    dim3 g1(T_ / 64, M3C_ / 128, B_);
    k_qkv<<<g1, 256, 0, stream>>>(wqh, xt, qkv_b, qkT, vh);

    dim3 g2(T_ / 64, B_ * HEADS_);
    k_attn<<<g2, 128, 0, stream>>>(qkT, vh, aT);

    dim3 g3(T_ / 64, C_ / 128, B_);
    k_proj<<<g3, 256, 0, stream>>>(wph, aT, proj_b, xn, rf);

    k_gn2<<<B_ * GROUPS_, 256, 0, stream>>>(rf, out);
}
